// _ItemModel_11613591569172
// MI455X (gfx1250) — matrix operand order repaired and checked by reading, not compiled, not run on hardware
//
#include <hip/hip_runtime.h>

// ---------------------------------------------------------------------------
// ItemModel (GraphRec-style item tower) for MI455X / gfx1250, wave32 + WMMA.
// Round 2: B weights pre-swizzled into WMMA-fragment-contiguous layout
// (2x b128 per fragment), vectorized LDS A loads, and hoisted q@W projections
// as batched WMMA GEMMs instead of per-workgroup scalar GEMVs.
//
// Input flattening assumption (setup_inputs() dict insertion order, recursive):
//   0 user_emb  1 item_emb  2 rate_emb
//   3 w1.w 4 w1.b 5 w2.w 6 w2.b 7 w3.w 8 w3.b 9 w4.w 10 w4.b
//   11 g_u.w1 12 g_u.b1 13 g_u.w2 14 g_u.b2
//   15..18 att_iu_i{w1,b1,w2,b2}  19..22 att_iu  23..26 att_if  27..30 att_iff
//   31 ag_users_i.w 32 .b  33 ag_users.w 34 .b  35 ag_if.w 36 .b  37 ag_iff.w 38 .b
//   39 combine.w1 40 combine.b1 41 combine.w2 42 combine.b2
//   43 uids 44 iids 45 i_user_pad 46 i_friends_pad 47 i_friends_user_pad
// ---------------------------------------------------------------------------

#define D_   64
#define B_   1024
#define U_   64
#define F_   32
#define S_   32
#define NU_  100000
#define NI_  50000
#define NR_  6
#define EPS_ 1e-10f

typedef __attribute__((ext_vector_type(16))) _Float16 v16h;
typedef __attribute__((ext_vector_type(8)))  _Float16 v8h;
typedef __attribute__((ext_vector_type(8)))  float    v8f;

// ---- WMMA fragment helpers (CDNA5 wave32 layouts, 16x16x32 f16) -----------

// A 16x32 (MxK), row-major source in LDS. Per lane the two 8-half runs are
// contiguous -> two ds_load_b128.
__device__ __forceinline__ v16h load_a16(const _Float16* A, int lda, int kBase, int lane) {
  const int half = (lane >> 4) & 1;
  const int m = lane & 15;
  const _Float16* p = A + m * lda + kBase + half * 8;
  const v8h lo = *(const v8h*)p;
  const v8h hi = *(const v8h*)(p + 16);
  v16h a;
#pragma unroll
  for (int e = 0; e < 8; ++e) { a[e] = lo[e]; a[e + 8] = hi[e]; }
  return a;
}

// B weights are pre-swizzled so that fragment (kt,nt), lane holds 16
// contiguous halfs: one aligned 32-byte load (2x global_load_b128).
// Packed index: ((kt*NT + nt)*32 + lane)*16 + e.
template<int KT, int NT>
__device__ __forceinline__ void gemm_tile(const _Float16* A, int lda,
                                          const _Float16* Bp,
                                          v8f* acc, int lane) {
#pragma unroll
  for (int kt = 0; kt < KT; ++kt) {
    v16h a = load_a16(A, lda, kt * 32, lane);
#pragma unroll
    for (int nt = 0; nt < NT; ++nt) {
      v16h b = *(const v16h*)(Bp + (size_t)((kt * NT + nt) * 32 + lane) * 16);
      acc[nt] = __builtin_amdgcn_wmma_f32_16x16x32_f16(
          false, a, false, b, (short)0, acc[nt], false, false);
    }
  }
}

// D layout: m = half*8 + r, n = lane&15.  act: 0=id, 1=leaky(0.2), 2=relu
template<int NT>
__device__ __forceinline__ void epilogue(const v8f* acc, const float* bias,
                                         _Float16* outH, int ldh,
                                         float* outF, int ldf,
                                         int act, int lane) {
  const int half = (lane >> 4) & 1;
  const int n = lane & 15;
#pragma unroll
  for (int nt = 0; nt < NT; ++nt) {
    const int c = nt * 16 + n;
    const float bv = bias ? bias[c] : 0.0f;
#pragma unroll
    for (int r = 0; r < 8; ++r) {
      const int m = half * 8 + r;
      float v = acc[nt][r] + bv;
      if (act == 1) v = (v > 0.0f) ? v : 0.2f * v;
      else if (act == 2) v = fmaxf(v, 0.0f);
      if (outH) outH[m * ldh + c] = (_Float16)v;
      if (outF) outF[m * ldf + c] = v;
    }
  }
}

// ---- fp32 -> fp16 conversion (embedding tables) ---------------------------
__global__ __launch_bounds__(256)
void f32_to_f16_kernel(const float* __restrict__ src, _Float16* __restrict__ dst, int n) {
  int i = blockIdx.x * blockDim.x + threadIdx.x;
  const int stride = gridDim.x * blockDim.x;
  for (; i < n; i += stride) dst[i] = (_Float16)src[i];
}

// ---- fp32 W[K][N] -> fragment-swizzled f16 --------------------------------
__global__ __launch_bounds__(256)
void pack_b_kernel(const float* __restrict__ src, _Float16* __restrict__ dst,
                   int K, int N) {
  const int NT = N >> 4;
  const int total = K * N;
  const int stride = gridDim.x * blockDim.x;
  for (int idx = blockIdx.x * blockDim.x + threadIdx.x; idx < total; idx += stride) {
    const int e    = idx & 15;
    const int lane = (idx >> 4) & 31;
    const int frag = idx >> 9;
    const int nt   = frag % NT;
    const int kt   = frag / NT;
    const int half = lane >> 4;
    const int n    = lane & 15;
    const int k    = kt * 32 + half * 8 + (e & 7) + ((e >> 3) << 4);
    dst[idx] = (_Float16)src[k * N + nt * 16 + n];
  }
}

// ---- batched q @ W + b projection (replaces per-workgroup GEMVs) ----------
// rows are gathered from item_emb_f16 via ids[row*idStride]; out fp32.
__global__ __launch_bounds__(64)
void proj_kernel(const _Float16* __restrict__ iembh,
                 const int* __restrict__ ids, int idStride,
                 const _Float16* __restrict__ Wp, const float* __restrict__ bias,
                 float* __restrict__ out) {
  __shared__ __align__(32) _Float16 sA[32][64];
  const int tid = threadIdx.x, lane = tid & 31, wave = tid >> 5, row0 = wave * 16;
  const int base = blockIdx.x * 32;
  {
    const int r = tid >> 1, h = tid & 1;
    const int id = ids[(base + r) * idStride];
    const v8h* s8 = (const v8h*)(iembh + id * D_ + h * 32);
    v8h* d8 = (v8h*)&sA[r][h * 32];
#pragma unroll
    for (int i = 0; i < 4; ++i) d8[i] = s8[i];
  }
  __syncthreads();
  v8f acc[4] = {};
  gemm_tile<2, 4>(&sA[row0][0], D_, Wp, acc, lane);
  epilogue<4>(acc, bias, nullptr, 0, out + (base + row0) * D_, D_, 0, lane);
}

// ---- block 1: users who rated the item (U=64 rows, 4 waves) ---------------
__global__ __launch_bounds__(128)
void block1_kernel(const _Float16* __restrict__ uembh,
                   const _Float16* __restrict__ rembh,
                   const int* __restrict__ i_user_pad,
                   const float* __restrict__ qw,       // [B][64] = q_i@w1+b1
                   const float* __restrict__ w1b,
                   const _Float16* __restrict__ w1p,
                   const _Float16* __restrict__ guw1p, const float* __restrict__ gub1,
                   const _Float16* __restrict__ guw2p, const float* __restrict__ gub2,
                   const _Float16* __restrict__ aw1p, const float* __restrict__ ab1,
                   const float* __restrict__ aw2, const float* __restrict__ ab2,
                   const float* __restrict__ agw, const float* __restrict__ agb,
                   float* __restrict__ z_out) {
  __shared__ __align__(32) _Float16 sA[64][128];
  __shared__ __align__(32) _Float16 sHid[64][64];
  __shared__ __align__(32) _Float16 sF[64][64];
  __shared__ float    sWF[64][64];
  __shared__ float    sQ[64];
  __shared__ float    sMask[64];
  __shared__ float    sScore[64];
  __shared__ float    sZ[64];
  __shared__ float    sRed;

  const int b    = blockIdx.x;
  const int tid  = threadIdx.x;
  const int lane = tid & 31;
  const int wave = tid >> 5;
  const int row0 = wave * 16;

  { // gather concat(user_emb, rate_emb) rows into f16 A (b128 copies)
    const int u = tid >> 1, h = tid & 1;
    const int uid = i_user_pad[(b * U_ + u) * 2 + 0];
    const int rid = i_user_pad[(b * U_ + u) * 2 + 1];
    if (h == 0) sMask[u] = (uid > 0) ? 1.0f : 0.0f;
    const v8h* s8 = (const v8h*)(h ? (rembh + rid * D_) : (uembh + uid * D_));
    v8h* d8 = (v8h*)&sA[u][h * D_];
#pragma unroll
    for (int i = 0; i < 8; ++i) d8[i] = s8[i];
  }
  if (tid < 64) sQ[tid] = qw[b * D_ + tid];
  __syncthreads();

  { // g_u layer 1: leaky_relu(A @ gw1 + gb1)
    v8f acc[4] = {};
    gemm_tile<4, 4>(&sA[row0][0], 128, guw1p, acc, lane);
    epilogue<4>(acc, gub1, &sHid[row0][0], D_, nullptr, 0, 1, lane);
  }
  { // g_u layer 2: f_jt = hid @ gw2 + gb2
    v8f acc[4] = {};
    gemm_tile<2, 4>(&sHid[row0][0], D_, guw2p, acc, lane);
    epilogue<4>(acc, gub2, &sF[row0][0], D_, nullptr, 0, 0, lane);
  }
  { // w1f = f_jt @ w1 + b1 -> att-A cols [0,64) (f16) and sWF (f32 for wsum)
    v8f acc[4] = {};
    gemm_tile<2, 4>(&sF[row0][0], D_, w1p, acc, lane);
    epilogue<4>(acc, w1b, &sA[row0][0], 128, &sWF[row0][0], D_, 0, lane);
  }
  { // att-A cols [64,128): mask ? (q_i@w1+b1) : b1
    const int u = row0 + (lane >> 1), h = lane & 1;
    const float mk = sMask[u];
#pragma unroll
    for (int i = 0; i < 32; ++i) {
      const int c = h * 32 + i;
      sA[u][64 + c] = (_Float16)((mk != 0.0f) ? sQ[c] : w1b[c]);
    }
  }
  { // attention MLP layer 1
    v8f acc[4] = {};
    gemm_tile<4, 4>(&sA[row0][0], 128, aw1p, acc, lane);
    epilogue<4>(acc, ab1, &sHid[row0][0], D_, nullptr, 0, 1, lane);
  }
  __syncthreads();
  if (tid < 64) { // attention layer 2 (N=1) + exp*mask
    float s = ab2[0];
    for (int k = 0; k < D_; ++k) s += (float)sHid[tid][k] * aw2[k];
    sScore[tid] = expf(s) * sMask[tid];
  }
  __syncthreads();
  if (tid < 32) { // softmax denominator (wave32 reduction)
    float v = sScore[tid] + sScore[tid + 32];
#pragma unroll
    for (int off = 16; off; off >>= 1) v += __shfl_xor(v, off, 32);
    if (lane == 0) sRed = v + EPS_;
  }
  __syncthreads();
  if (tid < 64) { // weighted sum over U of w1f
    const float inv = 1.0f / sRed;
    float z = 0.0f;
    for (int u = 0; u < U_; ++u) z += sScore[u] * inv * sWF[u][tid];
    sZ[tid] = z;
  }
  __syncthreads();
  if (tid < 64) { // aggre: relu(z @ agw + agb)
    float v = agb[tid];
    for (int k = 0; k < D_; ++k) v += sZ[k] * agw[k * D_ + tid];
    z_out[b * D_ + tid] = fmaxf(v, 0.0f);
  }
}

// ---- block 2: friend items of the item (F=32 rows, 2 waves) ---------------
__global__ __launch_bounds__(64)
void block2_kernel(const _Float16* __restrict__ iembh,
                   const int* __restrict__ i_friends_pad,
                   const float* __restrict__ qw,      // [B][64] = q_i@w2+b2
                   const float* __restrict__ w2b,
                   const _Float16* __restrict__ w2p,
                   const _Float16* __restrict__ aw1p, const float* __restrict__ ab1,
                   const float* __restrict__ aw2, const float* __restrict__ ab2,
                   const float* __restrict__ agw, const float* __restrict__ agb,
                   float* __restrict__ z_out) {
  __shared__ __align__(32) _Float16 sQa[32][64];
  __shared__ __align__(32) _Float16 sAtt[32][128];
  __shared__ __align__(32) _Float16 sHid[32][64];
  __shared__ float    sWA[32][64];
  __shared__ float    sQ[64];
  __shared__ float    sMask[32];
  __shared__ float    sScore[32];
  __shared__ float    sZ[64];
  __shared__ float    sRed;

  const int b    = blockIdx.x;
  const int tid  = threadIdx.x;
  const int lane = tid & 31;
  const int wave = tid >> 5;
  const int row0 = wave * 16;

  { // gather q_a = item_emb_f16[friend ids]
    const int u = tid >> 1, h = tid & 1;
    const int fid = i_friends_pad[(b * F_ + u) * 2 + 0];
    if (h == 0) sMask[u] = (fid > 0) ? 1.0f : 0.0f;
    const v8h* s8 = (const v8h*)(iembh + fid * D_ + h * 32);
    v8h* d8 = (v8h*)&sQa[u][h * 32];
#pragma unroll
    for (int i = 0; i < 4; ++i) d8[i] = s8[i];
  }
  sQ[tid] = qw[b * D_ + tid];
  __syncthreads();

  { // w2a = q_a @ w2 + b2
    v8f acc[4] = {};
    gemm_tile<2, 4>(&sQa[row0][0], D_, w2p, acc, lane);
    epilogue<4>(acc, w2b, &sAtt[row0][0], 128, &sWA[row0][0], D_, 0, lane);
  }
  { // att-A cols [64,128)
    const int u = row0 + (lane >> 1), h = lane & 1;
    const float mk = sMask[u];
#pragma unroll
    for (int i = 0; i < 32; ++i) {
      const int c = h * 32 + i;
      sAtt[u][64 + c] = (_Float16)((mk != 0.0f) ? sQ[c] : w2b[c]);
    }
  }
  { // attention MLP layer 1
    v8f acc[4] = {};
    gemm_tile<4, 4>(&sAtt[row0][0], 128, aw1p, acc, lane);
    epilogue<4>(acc, ab1, &sHid[row0][0], D_, nullptr, 0, 1, lane);
  }
  __syncthreads();
  if (tid < 32) {
    float s = ab2[0];
    for (int k = 0; k < D_; ++k) s += (float)sHid[tid][k] * aw2[k];
    sScore[tid] = expf(s) * sMask[tid];
  }
  __syncthreads();
  if (tid < 32) {
    float v = sScore[tid];
#pragma unroll
    for (int off = 16; off; off >>= 1) v += __shfl_xor(v, off, 32);
    if (lane == 0) sRed = v + EPS_;
  }
  __syncthreads();
  {
    const float inv = 1.0f / sRed;
    float z = 0.0f;
    for (int u = 0; u < F_; ++u) z += sScore[u] * inv * sWA[u][tid];
    sZ[tid] = z;
  }
  __syncthreads();
  {
    float v = agb[tid];
    for (int k = 0; k < D_; ++k) v += sZ[k] * agw[k * D_ + tid];
    z_out[b * D_ + tid] = fmaxf(v, 0.0f);
  }
}

// ---- block 3: users of each friend item (grid B*F, S=32 rows) -------------
__global__ __launch_bounds__(64)
void block3_kernel(const _Float16* __restrict__ uembh,
                   const _Float16* __restrict__ rembh,
                   const int* __restrict__ i_friends_user_pad,
                   const float* __restrict__ qw,      // [B*F][64] = q_a@w3+b3
                   const float* __restrict__ w3b,
                   const _Float16* __restrict__ w3p,
                   const _Float16* __restrict__ guw1p, const float* __restrict__ gub1,
                   const _Float16* __restrict__ guw2p, const float* __restrict__ gub2,
                   const _Float16* __restrict__ aw1p, const float* __restrict__ ab1,
                   const float* __restrict__ aw2, const float* __restrict__ ab2,
                   const float* __restrict__ agw, const float* __restrict__ agb,
                   float* __restrict__ h_out) {
  __shared__ __align__(32) _Float16 sA[32][128];
  __shared__ __align__(32) _Float16 sHid[32][64];
  __shared__ __align__(32) _Float16 sF[32][64];
  __shared__ float    sWX[32][64];
  __shared__ float    sQ[64];
  __shared__ float    sMask[32];
  __shared__ float    sScore[32];
  __shared__ float    sZ[64];
  __shared__ float    sRed;

  const int bf   = blockIdx.x;       // b*F + f
  const int tid  = threadIdx.x;
  const int lane = tid & 31;
  const int wave = tid >> 5;
  const int row0 = wave * 16;

  { // gather concat(user_emb, rate_emb) of the friend-item's raters
    const int srow = tid >> 1, h = tid & 1;
    const int uid = i_friends_user_pad[(bf * S_ + srow) * 2 + 0];
    const int rid = i_friends_user_pad[(bf * S_ + srow) * 2 + 1];
    if (h == 0) sMask[srow] = (uid > 0) ? 1.0f : 0.0f;
    const v8h* s8 = (const v8h*)(h ? (rembh + rid * D_) : (uembh + uid * D_));
    v8h* d8 = (v8h*)&sA[srow][h * D_];
#pragma unroll
    for (int i = 0; i < 8; ++i) d8[i] = s8[i];
  }
  sQ[tid] = qw[bf * D_ + tid];
  __syncthreads();

  { // g_u layer 1
    v8f acc[4] = {};
    gemm_tile<4, 4>(&sA[row0][0], 128, guw1p, acc, lane);
    epilogue<4>(acc, gub1, &sHid[row0][0], D_, nullptr, 0, 1, lane);
  }
  { // g_u layer 2 -> x_ia_s
    v8f acc[4] = {};
    gemm_tile<2, 4>(&sHid[row0][0], D_, guw2p, acc, lane);
    epilogue<4>(acc, gub2, &sF[row0][0], D_, nullptr, 0, 0, lane);
  }
  { // w3x = x @ w3 + b3
    v8f acc[4] = {};
    gemm_tile<2, 4>(&sF[row0][0], D_, w3p, acc, lane);
    epilogue<4>(acc, w3b, &sA[row0][0], 128, &sWX[row0][0], D_, 0, lane);
  }
  {
    const int u = row0 + (lane >> 1), h = lane & 1;
    const float mk = sMask[u];
#pragma unroll
    for (int i = 0; i < 32; ++i) {
      const int c = h * 32 + i;
      sA[u][64 + c] = (_Float16)((mk != 0.0f) ? sQ[c] : w3b[c]);
    }
  }
  { // att_iu layer 1
    v8f acc[4] = {};
    gemm_tile<4, 4>(&sA[row0][0], 128, aw1p, acc, lane);
    epilogue<4>(acc, ab1, &sHid[row0][0], D_, nullptr, 0, 1, lane);
  }
  __syncthreads();
  if (tid < 32) {
    float s = ab2[0];
    for (int k = 0; k < D_; ++k) s += (float)sHid[tid][k] * aw2[k];
    sScore[tid] = expf(s) * sMask[tid];
  }
  __syncthreads();
  if (tid < 32) {
    float v = sScore[tid];
#pragma unroll
    for (int off = 16; off; off >>= 1) v += __shfl_xor(v, off, 32);
    if (lane == 0) sRed = v + EPS_;
  }
  __syncthreads();
  {
    const float inv = 1.0f / sRed;
    float z = 0.0f;
    for (int s2 = 0; s2 < S_; ++s2) z += sScore[s2] * inv * sWX[s2][tid];
    sZ[tid] = z;
  }
  __syncthreads();
  { // h_oI = relu(z @ ag_users.w + b)
    float v = agb[tid];
    for (int k = 0; k < D_; ++k) v += sZ[k] * agw[k * D_ + tid];
    h_out[bf * D_ + tid] = fmaxf(v, 0.0f);
  }
}

// ---- block 4: aggregate friend summaries (F=32 rows per item) -------------
__global__ __launch_bounds__(64)
void block4_kernel(const float* __restrict__ h_oI,
                   const int* __restrict__ i_friends_pad,
                   const float* __restrict__ qw,      // [B][64] = q_i@w4+b4
                   const float* __restrict__ w4b,
                   const _Float16* __restrict__ w4p,
                   const _Float16* __restrict__ aw1p, const float* __restrict__ ab1,
                   const float* __restrict__ aw2, const float* __restrict__ ab2,
                   const float* __restrict__ agw, const float* __restrict__ agb,
                   float* __restrict__ z_out) {
  __shared__ __align__(32) _Float16 sHh[32][64];
  __shared__ float    sHf[32][64];
  __shared__ __align__(32) _Float16 sAtt[32][128];
  __shared__ __align__(32) _Float16 sHid[32][64];
  __shared__ float    sQ[64];
  __shared__ float    sMask[32];
  __shared__ float    sScore[32];
  __shared__ float    sZ[64];
  __shared__ float    sRed;

  const int b    = blockIdx.x;
  const int tid  = threadIdx.x;
  const int lane = tid & 31;
  const int wave = tid >> 5;
  const int row0 = wave * 16;

  { // stage h_oI rows (f32 for weighted sum, f16 for GEMM)
    const int u = tid >> 1, h = tid & 1;
    const int fid = i_friends_pad[(b * F_ + u) * 2 + 0];
    if (h == 0) sMask[u] = (fid > 0) ? 1.0f : 0.0f;
    const float4* s4 = (const float4*)(h_oI + (b * F_ + u) * D_ + h * 32);
#pragma unroll
    for (int i = 0; i < 8; ++i) {
      const float4 v = s4[i];
      const int c = h * 32 + i * 4;
      sHf[u][c + 0] = v.x; sHf[u][c + 1] = v.y;
      sHf[u][c + 2] = v.z; sHf[u][c + 3] = v.w;
      sHh[u][c + 0] = (_Float16)v.x; sHh[u][c + 1] = (_Float16)v.y;
      sHh[u][c + 2] = (_Float16)v.z; sHh[u][c + 3] = (_Float16)v.w;
    }
  }
  sQ[tid] = qw[b * D_ + tid];
  __syncthreads();

  { // w4h = h_oI @ w4 + b4
    v8f acc[4] = {};
    gemm_tile<2, 4>(&sHh[row0][0], D_, w4p, acc, lane);
    epilogue<4>(acc, w4b, &sAtt[row0][0], 128, nullptr, 0, 0, lane);
  }
  {
    const int u = row0 + (lane >> 1), h = lane & 1;
    const float mk = sMask[u];
#pragma unroll
    for (int i = 0; i < 32; ++i) {
      const int c = h * 32 + i;
      sAtt[u][64 + c] = (_Float16)((mk != 0.0f) ? sQ[c] : w4b[c]);
    }
  }
  { // att_iff layer 1
    v8f acc[4] = {};
    gemm_tile<4, 4>(&sAtt[row0][0], 128, aw1p, acc, lane);
    epilogue<4>(acc, ab1, &sHid[row0][0], D_, nullptr, 0, 1, lane);
  }
  __syncthreads();
  if (tid < 32) {
    float s = ab2[0];
    for (int k = 0; k < D_; ++k) s += (float)sHid[tid][k] * aw2[k];
    sScore[tid] = expf(s) * sMask[tid];
  }
  __syncthreads();
  if (tid < 32) {
    float v = sScore[tid];
#pragma unroll
    for (int off = 16; off; off >>= 1) v += __shfl_xor(v, off, 32);
    if (lane == 0) sRed = v + EPS_;
  }
  __syncthreads();
  { // weighted sum of h_oI (NOT w4h) per reference
    const float inv = 1.0f / sRed;
    float z = 0.0f;
    for (int u = 0; u < F_; ++u) z += sScore[u] * inv * sHf[u][tid];
    sZ[tid] = z;
  }
  __syncthreads();
  {
    float v = agb[tid];
    for (int k = 0; k < D_; ++k) v += sZ[k] * agw[k * D_ + tid];
    z_out[b * D_ + tid] = fmaxf(v, 0.0f);
  }
}

// ---- combine: 16 items per wave, 192->128->64 with relu -------------------
__global__ __launch_bounds__(32)
void combine_kernel(const float* __restrict__ z_if,
                    const float* __restrict__ z_j,
                    const float* __restrict__ z_uf,
                    const _Float16* __restrict__ cw1p, const float* __restrict__ cb1,
                    const _Float16* __restrict__ cw2p, const float* __restrict__ cb2,
                    float* __restrict__ out) {
  __shared__ __align__(32) _Float16 sZc[16][192];
  __shared__ __align__(32) _Float16 sH[16][128];

  const int ib   = blockIdx.x * 16;
  const int tid  = threadIdx.x;
  const int lane = tid & 31;

  for (int idx = tid; idx < 16 * 192; idx += 32) {
    const int row = idx / 192, col = idx % 192;
    const int item = ib + row;
    float v;
    if (col < 64)       v = z_if[item * D_ + col];
    else if (col < 128) v = z_j[item * D_ + (col - 64)];
    else                v = z_uf[item * D_ + (col - 128)];
    sZc[row][col] = (_Float16)v;
  }
  __syncthreads();
  { // layer 1: relu(z @ cw1 + cb1), K=192, N=128
    v8f acc[8] = {};
    gemm_tile<6, 8>(&sZc[0][0], 192, cw1p, acc, lane);
    epilogue<8>(acc, cb1, &sH[0][0], 128, nullptr, 0, 2, lane);
  }
  __syncthreads();
  { // layer 2: relu(h @ cw2 + cb2), K=128, N=64 -> output (fp32)
    v8f acc[4] = {};
    gemm_tile<4, 4>(&sH[0][0], 128, cw2p, acc, lane);
    epilogue<4>(acc, cb2, nullptr, 0, out + ib * D_, D_, 2, lane);
  }
}

// ---------------------------------------------------------------------------
extern "C" void kernel_launch(void* const* d_in, const int* in_sizes, int n_in,
                              void* d_out, int out_size, void* d_ws, size_t ws_size,
                              hipStream_t stream) {
  (void)in_sizes; (void)n_in; (void)out_size; (void)ws_size;

  enum {
    IN_USER_EMB = 0, IN_ITEM_EMB, IN_RATE_EMB,
    IN_W1_W, IN_W1_B, IN_W2_W, IN_W2_B, IN_W3_W, IN_W3_B, IN_W4_W, IN_W4_B,
    IN_GU_W1, IN_GU_B1, IN_GU_W2, IN_GU_B2,
    IN_AIUI_W1, IN_AIUI_B1, IN_AIUI_W2, IN_AIUI_B2,
    IN_AIU_W1, IN_AIU_B1, IN_AIU_W2, IN_AIU_B2,
    IN_AIF_W1, IN_AIF_B1, IN_AIF_W2, IN_AIF_B2,
    IN_AIFF_W1, IN_AIFF_B1, IN_AIFF_W2, IN_AIFF_B2,
    IN_AGUI_W, IN_AGUI_B, IN_AGU_W, IN_AGU_B,
    IN_AGIF_W, IN_AGIF_B, IN_AGIFF_W, IN_AGIFF_B,
    IN_COMB_W1, IN_COMB_B1, IN_COMB_W2, IN_COMB_B2,
    IN_UIDS, IN_IIDS, IN_I_USER_PAD, IN_I_FRIENDS_PAD, IN_I_FRIENDS_USER_PAD
  };
  #define FP(i) ((const float*)d_in[(i)])
  #define IP(i) ((const int*)d_in[(i)])

  // ---- workspace bump allocator ----
  char* wp = (char*)d_ws;
  auto alloc = [&](size_t bytes) -> void* {
    void* p = (void*)wp;
    wp += (bytes + 255) & ~(size_t)255;
    return p;
  };
  _Float16* uembh  = (_Float16*)alloc((size_t)NU_ * D_ * 2);
  _Float16* iembh  = (_Float16*)alloc((size_t)NI_ * D_ * 2);
  _Float16* rembh  = (_Float16*)alloc((size_t)NR_ * D_ * 2);
  _Float16* w1p    = (_Float16*)alloc(D_ * D_ * 2);
  _Float16* w2p    = (_Float16*)alloc(D_ * D_ * 2);
  _Float16* w3p    = (_Float16*)alloc(D_ * D_ * 2);
  _Float16* w4p    = (_Float16*)alloc(D_ * D_ * 2);
  _Float16* guw1p  = (_Float16*)alloc(2 * D_ * D_ * 2);
  _Float16* guw2p  = (_Float16*)alloc(D_ * D_ * 2);
  _Float16* aiui1p = (_Float16*)alloc(2 * D_ * D_ * 2);
  _Float16* aiu1p  = (_Float16*)alloc(2 * D_ * D_ * 2);
  _Float16* aif1p  = (_Float16*)alloc(2 * D_ * D_ * 2);
  _Float16* aiff1p = (_Float16*)alloc(2 * D_ * D_ * 2);
  _Float16* cw1p   = (_Float16*)alloc(3 * D_ * 2 * D_ * 2);
  _Float16* cw2p   = (_Float16*)alloc(2 * D_ * D_ * 2);
  float* z_j  = (float*)alloc((size_t)B_ * D_ * 4);
  float* z_if = (float*)alloc((size_t)B_ * D_ * 4);
  float* z_uf = (float*)alloc((size_t)B_ * D_ * 4);
  float* h_oI = (float*)alloc((size_t)B_ * F_ * D_ * 4);
  float* qw1  = (float*)alloc((size_t)B_ * D_ * 4);
  float* qw2  = (float*)alloc((size_t)B_ * D_ * 4);
  float* qw4  = (float*)alloc((size_t)B_ * D_ * 4);
  float* qw3  = (float*)alloc((size_t)B_ * F_ * D_ * 4);

  // ---- fp16 embedding tables ----
  auto cvt = [&](const void* src, _Float16* dst, int n) {
    int blocks = (n + 255) / 256;
    if (blocks > 4096) blocks = 4096;
    f32_to_f16_kernel<<<blocks, 256, 0, stream>>>((const float*)src, dst, n);
  };
  cvt(d_in[IN_USER_EMB], uembh, NU_ * D_);
  cvt(d_in[IN_ITEM_EMB], iembh, NI_ * D_);
  cvt(d_in[IN_RATE_EMB], rembh, NR_ * D_);

  // ---- fragment-swizzled f16 weights (stay hot in 192MB L2) ----
  auto pack = [&](const void* src, _Float16* dst, int K, int N) {
    int n = K * N;
    int blocks = (n + 255) / 256;
    if (blocks > 1024) blocks = 1024;
    pack_b_kernel<<<blocks, 256, 0, stream>>>((const float*)src, dst, K, N);
  };
  pack(d_in[IN_W1_W], w1p, D_, D_);
  pack(d_in[IN_W2_W], w2p, D_, D_);
  pack(d_in[IN_W3_W], w3p, D_, D_);
  pack(d_in[IN_W4_W], w4p, D_, D_);
  pack(d_in[IN_GU_W1], guw1p, 2 * D_, D_);
  pack(d_in[IN_GU_W2], guw2p, D_, D_);
  pack(d_in[IN_AIUI_W1], aiui1p, 2 * D_, D_);
  pack(d_in[IN_AIU_W1], aiu1p, 2 * D_, D_);
  pack(d_in[IN_AIF_W1], aif1p, 2 * D_, D_);
  pack(d_in[IN_AIFF_W1], aiff1p, 2 * D_, D_);
  pack(d_in[IN_COMB_W1], cw1p, 3 * D_, 2 * D_);
  pack(d_in[IN_COMB_W2], cw2p, 2 * D_, D_);

  // ---- hoisted q @ W + b projections (batched WMMA GEMMs) ----
  proj_kernel<<<B_ / 32, 64, 0, stream>>>(iembh, IP(IN_IIDS), 1, w1p, FP(IN_W1_B), qw1);
  proj_kernel<<<B_ / 32, 64, 0, stream>>>(iembh, IP(IN_IIDS), 1, w2p, FP(IN_W2_B), qw2);
  proj_kernel<<<B_ / 32, 64, 0, stream>>>(iembh, IP(IN_IIDS), 1, w4p, FP(IN_W4_B), qw4);
  proj_kernel<<<(B_ * F_) / 32, 64, 0, stream>>>(iembh, IP(IN_I_FRIENDS_PAD), 2,
                                                 w3p, FP(IN_W3_B), qw3);

  // ---- block 1 ----
  block1_kernel<<<B_, 128, 0, stream>>>(
      uembh, rembh, IP(IN_I_USER_PAD), qw1, FP(IN_W1_B), w1p,
      guw1p, FP(IN_GU_B1), guw2p, FP(IN_GU_B2),
      aiui1p, FP(IN_AIUI_B1), FP(IN_AIUI_W2), FP(IN_AIUI_B2),
      FP(IN_AGUI_W), FP(IN_AGUI_B), z_j);

  // ---- block 2 ----
  block2_kernel<<<B_, 64, 0, stream>>>(
      iembh, IP(IN_I_FRIENDS_PAD), qw2, FP(IN_W2_B), w2p,
      aif1p, FP(IN_AIF_B1), FP(IN_AIF_W2), FP(IN_AIF_B2),
      FP(IN_AGIF_W), FP(IN_AGIF_B), z_if);

  // ---- block 3 (dominant; one workgroup per (item, friend)) ----
  block3_kernel<<<B_ * F_, 64, 0, stream>>>(
      uembh, rembh, IP(IN_I_FRIENDS_USER_PAD), qw3, FP(IN_W3_B), w3p,
      guw1p, FP(IN_GU_B1), guw2p, FP(IN_GU_B2),
      aiu1p, FP(IN_AIU_B1), FP(IN_AIU_W2), FP(IN_AIU_B2),
      FP(IN_AGU_W), FP(IN_AGU_B), h_oI);

  // ---- block 4 ----
  block4_kernel<<<B_, 64, 0, stream>>>(
      h_oI, IP(IN_I_FRIENDS_PAD), qw4, FP(IN_W4_B), w4p,
      aiff1p, FP(IN_AIFF_B1), FP(IN_AIFF_W2), FP(IN_AIFF_B2),
      FP(IN_AGIFF_W), FP(IN_AGIFF_B), z_uf);

  // ---- combine ----
  combine_kernel<<<B_ / 16, 32, 0, stream>>>(
      z_if, z_j, z_uf, cw1p, FP(IN_COMB_B1), cw2p, FP(IN_COMB_B2),
      (float*)d_out);

  #undef FP
  #undef IP
}
